// CausalSelfAttention_13649406066848
// MI455X (gfx1250) — compile-verified, matrix operand order checked by reading, not run on hardware
//
#include <hip/hip_runtime.h>

typedef unsigned short u16;
typedef __attribute__((ext_vector_type(8)))  u16    u16x8;
typedef __attribute__((ext_vector_type(16))) __bf16 bf16x16;
typedef __attribute__((ext_vector_type(8)))  float  f32x8;
typedef int i32x4 __attribute__((vector_size(16)));   // matches builtin's V4i param

#define Bsz 2
#define Tsz 4096
#define Csz 1024
#define Hsz 16
#define HDsz 64

// ---------- CDNA5 async global->LDS copy (ASYNCcnt path) ----------

#if __has_builtin(__builtin_amdgcn_global_load_async_to_lds_b128)
#define HAS_ASYNC_LDS 1
#else
#define HAS_ASYNC_LDS 0
#endif

__device__ __forceinline__ void async_cp16(const u16* g, u16* l) {
#if HAS_ASYNC_LDS
    __builtin_amdgcn_global_load_async_to_lds_b128(
        (__attribute__((address_space(1))) i32x4*)g,
        (__attribute__((address_space(3))) i32x4*)l,
        0, 0);
#else
    *(u16x8*)l = *(const u16x8*)g;   // fallback: load+ds_store (synchronous)
#endif
}

__device__ __forceinline__ void async_wait0() {
#if HAS_ASYNC_LDS
#if __has_builtin(__builtin_amdgcn_s_wait_asynccnt)
    __builtin_amdgcn_s_wait_asynccnt(0);
#else
    asm volatile("s_wait_asynccnt 0x0" ::: "memory");
#endif
#endif
}

// ---------- helpers ----------

__device__ __forceinline__ u16 f2bf(float f) {  // round-to-nearest-even fp32 -> bf16
    unsigned u = __float_as_uint(f);
    u += 0x7FFFu + ((u >> 16) & 1u);
    return (u16)(u >> 16);
}

union FragCvt { u16x8 u[2]; bf16x16 b; };

// Load a 16x32 bf16 A-frag (or symmetric B-frag) for one lane.
// p points at tile_row[lane%16] + kbase + (lane/16)*8 ; second chunk is +16 elems.
__device__ __forceinline__ bf16x16 frag_ld(const u16* p) {
    FragCvt f;
    f.u[0] = *(const u16x8*)p;
    f.u[1] = *(const u16x8*)(p + 16);
    return f.b;
}

__device__ __forceinline__ f32x8 wmma_bf16(bf16x16 a, bf16x16 b, f32x8 c) {
    return __builtin_amdgcn_wmma_f32_16x16x32_bf16(false, a, false, b, (short)0, c, false, false);
}

// ---------- stage 1: conversions ----------

__global__ void convert_bf16(const float* __restrict__ src, u16* __restrict__ dst, size_t n) {
    size_t i = (size_t)blockIdx.x * 256 + threadIdx.x;
    if (i < n) dst[i] = f2bf(src[i]);
}

// src: K x N fp32 row-major ; dst: N x K bf16 row-major (transposed)
__global__ void transpose_bf16(const float* __restrict__ src, u16* __restrict__ dst, int K, int N) {
    size_t i = (size_t)blockIdx.x * 256 + threadIdx.x;
    if (i >= (size_t)K * N) return;
    int    k = (int)(i % (size_t)K);
    size_t n = i / (size_t)K;
    dst[n * (size_t)K + k] = f2bf(src[(size_t)k * N + n]);
}

// ---------- stage 2/5: bf16 GEMM  C[M,N] = A[M,K] * Bt[N,K]^T + bias ----------
// Double-buffered LDS tiles filled with async global->LDS copies.

#define BM 128
#define BN 128
#define BK 32

__global__ __launch_bounds__(256) void gemm_bf16_nt(
    const u16*   __restrict__ A,     // M x K (bf16 bits)
    const u16*   __restrict__ Bt,    // N x K (bf16 bits, pre-transposed)
    const float* __restrict__ bias,  // N
    float*       __restrict__ Cout,  // M x N fp32
    int M, int N, int K)
{
    __shared__ __align__(16) u16 As[2][BM][BK];
    __shared__ __align__(16) u16 Bs[2][BN][BK];

    const int tid  = threadIdx.x;
    const int lane = tid & 31;
    const int wave = tid >> 5;            // 0..7
    const int wm   = (wave >> 2) * 64;    // 0 / 64
    const int wn   = (wave & 3) * 32;     // 0 / 32 / 64 / 96
    const int half = lane >> 4;           // 0 / 1
    const int l16  = lane & 15;
    const int m0   = blockIdx.y * BM;
    const int n0   = blockIdx.x * BN;

    auto stage = [&](int buf, int k0) {
#pragma unroll
        for (int i = 0; i < 2; ++i) {
            int c   = tid + i * 256;
            int row = c >> 2;
            int kc  = (c & 3) * 8;
            async_cp16(&A [(size_t)(m0 + row) * K + k0 + kc], &As[buf][row][kc]);
            async_cp16(&Bt[(size_t)(n0 + row) * K + k0 + kc], &Bs[buf][row][kc]);
        }
    };

    f32x8 acc[4][2];
#pragma unroll
    for (int i = 0; i < 4; ++i)
#pragma unroll
        for (int j = 0; j < 2; ++j)
            acc[i][j] = (f32x8){0.f,0.f,0.f,0.f,0.f,0.f,0.f,0.f};

    stage(0, 0);
    int buf = 0;
    for (int k0 = 0; k0 < K; k0 += BK) {
        async_wait0();       // this wave's async copies into `buf` are done
        __syncthreads();     // everyone's copies done; everyone done reading buf^1
        if (k0 + BK < K) stage(buf ^ 1, k0 + BK);   // overlap next tile with math

        bf16x16 bfrag[2];
#pragma unroll
        for (int j = 0; j < 2; ++j)
            bfrag[j] = frag_ld(&Bs[buf][wn + j * 16 + l16][half * 8]);
#pragma unroll
        for (int i = 0; i < 4; ++i) {
            bf16x16 afrag = frag_ld(&As[buf][wm + i * 16 + l16][half * 8]);
#pragma unroll
            for (int j = 0; j < 2; ++j)
                acc[i][j] = wmma_bf16(afrag, bfrag[j], acc[i][j]);
        }
        buf ^= 1;
    }

    // C/D layout: reg r <-> row (lane/16)*8 + r, col lane%16
#pragma unroll
    for (int i = 0; i < 4; ++i)
#pragma unroll
        for (int j = 0; j < 2; ++j) {
            int mrow = m0 + wm + i * 16 + half * 8;
            int ncol = n0 + wn + j * 16 + l16;
            float bv = bias[ncol];
#pragma unroll
            for (int r = 0; r < 8; ++r)
                Cout[(size_t)(mrow + r) * N + ncol] = acc[i][j][r] + bv;
        }
}

// ---------- stage 3: split qkv, reshape to heads, emit k/v fp32 + bf16 (V transposed) ----------

__global__ void scatter_qkv(const float* __restrict__ qkv,   // (B,T,3C)
                            float* __restrict__ kout,        // (B,H,T,HD) fp32 -> d_out
                            float* __restrict__ vout,        // (B,H,T,HD) fp32 -> d_out
                            u16*   __restrict__ qb,          // (B,H,T,HD) bf16
                            u16*   __restrict__ kb,          // (B,H,T,HD) bf16
                            u16*   __restrict__ vt)          // (B,H,HD,T) bf16 (transposed)
{
    size_t i = (size_t)blockIdx.x * 256 + threadIdx.x;   // over B*T*C
    int    c  = (int)(i & (Csz - 1));
    size_t tb = i >> 10;
    int    t  = (int)(tb & (Tsz - 1));
    int    b  = (int)(tb >> 12);
    int    h  = c >> 6, d = c & (HDsz - 1);

    size_t qbase = ((size_t)b * Tsz + t) * (3 * Csz);
    float q = qkv[qbase + c];
    float k = qkv[qbase + Csz + c];
    float v = qkv[qbase + 2 * Csz + c];

    size_t hidx = (((size_t)(b * Hsz + h) * Tsz) + t) * HDsz + d;
    kout[hidx] = k;
    vout[hidx] = v;
    qb[hidx]   = f2bf(q);
    kb[hidx]   = f2bf(k);
    vt[((size_t)(b * Hsz + h) * HDsz + d) * Tsz + t] = f2bf(v);
}

// ---------- stage 4: causal flash attention, 64-query tile, 64-key blocks ----------
// Double-buffered K/V tiles via async global->LDS copies.

__global__ __launch_bounds__(128) void attn_fwd(
    const u16* __restrict__ qb,   // (B*H, T, 64)
    const u16* __restrict__ kb,   // (B*H, T, 64)
    const u16* __restrict__ vt,   // (B*H, 64, T)
    u16*       __restrict__ yb)   // (B, T, C) bf16
{
    __shared__ __align__(16) u16 Qs[64][64];
    __shared__ __align__(16) u16 Ks[2][64][64];
    __shared__ __align__(16) u16 Vs[2][64][64];   // [d][key]
    __shared__ __align__(16) u16 Ps[64][64];      // wave-private 16-row bands

    const int bh   = blockIdx.y;            // b*H + h
    const int b    = bh >> 4, h = bh & 15;
    const int qt   = blockIdx.x;            // query tile
    const int tid  = threadIdx.x;
    const int lane = tid & 31;
    const int wave = tid >> 5;              // 4 waves, 16 q-rows each
    const int half = lane >> 4;
    const int l16  = lane & 15;

    const size_t base = (size_t)bh * Tsz * HDsz;

    auto stageKV = [&](int buf, int jb) {
#pragma unroll
        for (int i = 0; i < 4; ++i) {
            int c = tid + i * 128, row = c >> 3, kc = (c & 7) * 8;
            async_cp16(&kb[base + (size_t)(jb * 64 + row) * HDsz + kc], &Ks[buf][row][kc]);
            async_cp16(&vt[base + (size_t)row * Tsz + jb * 64 + kc],    &Vs[buf][row][kc]);
        }
    };

    // stage Q tile + first K/V block together
#pragma unroll
    for (int i = 0; i < 4; ++i) {
        int c = tid + i * 128, row = c >> 3, kc = (c & 7) * 8;
        async_cp16(&qb[base + (size_t)(qt * 64 + row) * HDsz + kc], &Qs[row][kc]);
    }
    stageKV(0, 0);
    async_wait0();
    __syncthreads();

    bf16x16 qfrag[2];
#pragma unroll
    for (int kt = 0; kt < 2; ++kt)
        qfrag[kt] = frag_ld(&Qs[wave * 16 + l16][kt * 32 + half * 8]);

    f32x8 oacc[4];
#pragma unroll
    for (int dt = 0; dt < 4; ++dt) oacc[dt] = (f32x8){0.f,0.f,0.f,0.f,0.f,0.f,0.f,0.f};
    float rmax[8], rsum[8];
#pragma unroll
    for (int r = 0; r < 8; ++r) { rmax[r] = -1e30f; rsum[r] = 0.f; }

    const int   qrow0 = qt * 64 + wave * 16 + half * 8;  // + r => this lane's rows
    const float scale = 0.125f;                          // 1/sqrt(64)
    const int   nkv   = qt + 1;                          // causal bound

    int buf = 0;
    for (int jb = 0; jb < nkv; ++jb) {
        if (jb) {            // first block already waited above
            async_wait0();
            __syncthreads();
        }
        if (jb + 1 < nkv) stageKV(buf ^ 1, jb + 1);   // overlap next K/V with math

        // S = Q K^T  (B-frag = rows of K tile, K-contiguous)
        f32x8 s[4];
#pragma unroll
        for (int nt = 0; nt < 4; ++nt) {
            f32x8 z = (f32x8){0.f,0.f,0.f,0.f,0.f,0.f,0.f,0.f};
            z     = wmma_bf16(qfrag[0], frag_ld(&Ks[buf][nt * 16 + l16][half * 8]),      z);
            s[nt] = wmma_bf16(qfrag[1], frag_ld(&Ks[buf][nt * 16 + l16][32 + half * 8]), z);
        }

        // scale + causal mask + running row max
        float nmax[8];
#pragma unroll
        for (int r = 0; r < 8; ++r) nmax[r] = rmax[r];
#pragma unroll
        for (int nt = 0; nt < 4; ++nt) {
            int key = jb * 64 + nt * 16 + l16;
#pragma unroll
            for (int r = 0; r < 8; ++r) {
                float v = s[nt][r] * scale;
                if (key > qrow0 + r) v = -1e30f;
                s[nt][r] = v;
                nmax[r] = fmaxf(nmax[r], v);
            }
        }
#pragma unroll
        for (int off = 1; off < 16; off <<= 1)
#pragma unroll
            for (int r = 0; r < 8; ++r)
                nmax[r] = fmaxf(nmax[r], __shfl_xor(nmax[r], off, 32));

        float corr[8], psum[8];
#pragma unroll
        for (int r = 0; r < 8; ++r) {
            corr[r] = __expf(rmax[r] - nmax[r]);
            rmax[r] = nmax[r];
            psum[r] = 0.f;
        }
#pragma unroll
        for (int nt = 0; nt < 4; ++nt)
#pragma unroll
            for (int r = 0; r < 8; ++r) {
                float p = __expf(s[nt][r] - rmax[r]);
                s[nt][r] = p;
                psum[r] += p;
            }
#pragma unroll
        for (int off = 1; off < 16; off <<= 1)
#pragma unroll
            for (int r = 0; r < 8; ++r)
                psum[r] += __shfl_xor(psum[r], off, 32);
#pragma unroll
        for (int r = 0; r < 8; ++r) rsum[r] = rsum[r] * corr[r] + psum[r];
#pragma unroll
        for (int dt = 0; dt < 4; ++dt)
#pragma unroll
            for (int r = 0; r < 8; ++r) oacc[dt][r] *= corr[r];

        // C-layout -> A-layout via wave-private LDS band
#pragma unroll
        for (int nt = 0; nt < 4; ++nt)
#pragma unroll
            for (int r = 0; r < 8; ++r)
                Ps[wave * 16 + half * 8 + r][nt * 16 + l16] = f2bf(s[nt][r]);

        bf16x16 pf0 = frag_ld(&Ps[wave * 16 + l16][half * 8]);
        bf16x16 pf1 = frag_ld(&Ps[wave * 16 + l16][32 + half * 8]);
#pragma unroll
        for (int dt = 0; dt < 4; ++dt) {
            oacc[dt] = wmma_bf16(pf0, frag_ld(&Vs[buf][dt * 16 + l16][half * 8]),      oacc[dt]);
            oacc[dt] = wmma_bf16(pf1, frag_ld(&Vs[buf][dt * 16 + l16][32 + half * 8]), oacc[dt]);
        }
        buf ^= 1;
    }

    // normalize and emit in (B,T,C) bf16 layout for the projection GEMM
#pragma unroll
    for (int r = 0; r < 8; ++r) rsum[r] = 1.0f / rsum[r];
#pragma unroll
    for (int dt = 0; dt < 4; ++dt) {
        int d = dt * 16 + l16;
#pragma unroll
        for (int r = 0; r < 8; ++r) {
            int t = qrow0 + r;
            yb[((size_t)(b * Tsz + t)) * Csz + h * HDsz + d] = f2bf(oacc[dt][r] * rsum[r]);
        }
    }
}

// ---------- host orchestration ----------

extern "C" void kernel_launch(void* const* d_in, const int* in_sizes, int n_in,
                              void* d_out, int out_size, void* d_ws, size_t ws_size,
                              hipStream_t stream) {
    (void)in_sizes; (void)n_in; (void)out_size; (void)ws_size;
    const float* x      = (const float*)d_in[0];
    const float* w_attn = (const float*)d_in[1];
    const float* b_attn = (const float*)d_in[2];
    const float* w_proj = (const float*)d_in[3];
    const float* b_proj = (const float*)d_in[4];
    float* out = (float*)d_out;

    const size_t M  = (size_t)Bsz * Tsz;   // 8192
    const size_t MC = M * Csz;             // 8,388,608

    char* ws  = (char*)d_ws;
    size_t off = 0;
    auto carve = [&](size_t bytes) { void* p = ws + off; off += (bytes + 255) & ~(size_t)255; return p; };

    float* qkv = (float*)carve(M * 3 * Csz * sizeof(float));   // 96 MB
    u16*   xb  = (u16*)  carve(MC * sizeof(u16));
    u16*   wAt = (u16*)  carve((size_t)3 * Csz * Csz * sizeof(u16));
    u16*   wPt = (u16*)  carve((size_t)Csz * Csz * sizeof(u16));
    u16*   qb  = (u16*)  carve(MC * sizeof(u16));
    u16*   kb  = (u16*)  carve(MC * sizeof(u16));
    u16*   vt  = (u16*)  carve(MC * sizeof(u16));
    u16*   yb  = xb;  // x-bf16 is dead after the QKV GEMM; reuse for attention output

    float* kout = out + MC;       // d_out layout: [y | k | v]
    float* vout = out + 2 * MC;

    // 1) fp32 -> bf16 conversions (weights pre-transposed to N x K)
    convert_bf16 <<<dim3((unsigned)(MC / 256)), dim3(256), 0, stream>>>(x, xb, MC);
    transpose_bf16<<<dim3((unsigned)((3 * Csz * Csz + 255) / 256)), dim3(256), 0, stream>>>(w_attn, wAt, Csz, 3 * Csz);
    transpose_bf16<<<dim3((unsigned)((Csz * Csz + 255) / 256)),     dim3(256), 0, stream>>>(w_proj, wPt, Csz, Csz);

    // 2) QKV projection: (8192 x 1024) @ (1024 x 3072) + b_attn
    gemm_bf16_nt<<<dim3(3 * Csz / BN, (unsigned)(M / BM)), dim3(256), 0, stream>>>(
        xb, wAt, b_attn, qkv, (int)M, 3 * Csz, Csz);

    // 3) split/reshape; k,v fp32 into d_out; bf16 q/k + transposed v for attention
    scatter_qkv<<<dim3((unsigned)(MC / 256)), dim3(256), 0, stream>>>(qkv, kout, vout, qb, kb, vt);

    // 4) causal attention
    attn_fwd<<<dim3(Tsz / 64, Bsz * Hsz), dim3(128), 0, stream>>>(qb, kb, vt, yb);

    // 5) output projection: (8192 x 1024) @ (1024 x 1024) + b_proj -> y
    gemm_bf16_nt<<<dim3(Csz / BN, (unsigned)(M / BM)), dim3(256), 0, stream>>>(
        yb, wPt, b_proj, out, (int)M, Csz, Csz);
}
